// MultiheadAttention_82308753260917
// MI455X (gfx1250) — compile-verified, matrix operand order checked by reading, not run on hardware
//
#include <hip/hip_runtime.h>

// ---------------------------------------------------------------------------
// CDNA5 (gfx1250) multihead-attention forward, bf16 WMMA everywhere.
// wave32; v_wmma_f32_16x16x32_bf16; async-to-LDS double-buffered K/V staging.
// ---------------------------------------------------------------------------

typedef __attribute__((ext_vector_type(16))) __bf16 v16bf;
typedef __attribute__((ext_vector_type(8)))  float  v8f;
typedef __attribute__((ext_vector_type(4)))  int    v4i;

struct alignas(16) U4 { unsigned int x, y, z, w; };
struct alignas(16) F4 { float x, y, z, w; };

union Frag { v16bf v; U4 q[2]; };

// ---- CDNA5 async global->LDS copy (ASYNCcnt), guarded with fallback --------
#if defined(__AMDGCN__) && __has_builtin(__builtin_amdgcn_global_load_async_to_lds_b128)
#define HAVE_ASYNC_LDS 1
#endif

#define AS_GLOBAL __attribute__((address_space(1)))
#define AS_LDS    __attribute__((address_space(3)))

__device__ __forceinline__ void copy16_g2l(const __bf16* g, __bf16* l)
{
#ifdef HAVE_ASYNC_LDS
    __builtin_amdgcn_global_load_async_to_lds_b128(
        (AS_GLOBAL v4i*)g, (AS_LDS v4i*)l, 0, 0);
#else
    *(U4*)l = *(const U4*)g;
#endif
}

__device__ __forceinline__ void async_wait_le4()
{
#ifdef HAVE_ASYNC_LDS
#if __has_builtin(__builtin_amdgcn_s_wait_asynccnt)
    __builtin_amdgcn_s_wait_asynccnt(4);
#else
    asm volatile("s_wait_asynccnt 0x4" ::: "memory");
#endif
#endif
}

__device__ __forceinline__ void async_wait_0()
{
#ifdef HAVE_ASYNC_LDS
#if __has_builtin(__builtin_amdgcn_s_wait_asynccnt)
    __builtin_amdgcn_s_wait_asynccnt(0);
#else
    asm volatile("s_wait_asynccnt 0x0" ::: "memory");
#endif
#endif
}

// ---------------------------------------------------------------------------
// Generic GEMM:  C[M,N] = A[M,K] * B[K,N] + bias[N]
//   A fp32 (converted to bf16 while staging) or already bf16.
//   Block tile 128x128, K-step 32, 256 threads = 8 waves, wave tile 32x64.
//   Bs staged transposed [n][k] so B fragments are contiguous b128 LDS loads.
// ---------------------------------------------------------------------------
template <bool A_BF16>
__global__ __launch_bounds__(256)
void gemm_bf16_wmma(const void* __restrict__ Ap,
                    const float* __restrict__ B,
                    const float* __restrict__ bias,
                    float* __restrict__ C,
                    int M, int N, int K)
{
    __shared__ __bf16 As[128][40];   // pitch 40 bf16 = 80B (16B-aligned rows)
    __shared__ __bf16 Bs[128][40];   // [n][k]

    const int tid  = threadIdx.x;
    const int lane = tid & 31;
    const int wave = tid >> 5;
    const int wm   = wave >> 1;                  // 0..3 (M groups of 32)
    const int wn   = wave & 1;                   // 0..1 (N groups of 64)
    const int m0   = blockIdx.y * 128;
    const int n0   = blockIdx.x * 128;
    const int ln   = lane & 15;
    const int klo  = (lane < 16) ? 0 : 8;        // K-group per ISA A/B layout
    const int rlo  = (lane < 16) ? 0 : 8;        // C-layout row offset

    const v8f vz = {0.f,0.f,0.f,0.f,0.f,0.f,0.f,0.f};
    v8f acc[2][4];
#pragma unroll
    for (int i = 0; i < 2; ++i)
#pragma unroll
        for (int j = 0; j < 4; ++j) acc[i][j] = vz;

    const int arow  = tid >> 1;                  // 0..127
    const int ahalf = (tid & 1) * 16;            // 0 or 16 (K columns)
    const int bk    = tid & 31;                  // 0..31  (K row of B)
    const int bn    = (tid >> 5) * 16;           // 0..112 (N column group)

    for (int kk = 0; kk < K; kk += 32) {
        // ---- stage A tile (128x32) as bf16 ----
        if (A_BF16) {
            const __bf16* A = (const __bf16*)Ap;
            const U4* src = (const U4*)(A + (size_t)(m0 + arow) * K + kk + ahalf);
            *(U4*)&As[arow][ahalf]     = src[0];
            *(U4*)&As[arow][ahalf + 8] = src[1];
        } else {
            const float* A = (const float*)Ap;
            const F4* src = (const F4*)(A + (size_t)(m0 + arow) * K + kk + ahalf);
#pragma unroll
            for (int i = 0; i < 4; ++i) {
                F4 f = src[i];
                As[arow][ahalf + 4*i + 0] = (__bf16)f.x;
                As[arow][ahalf + 4*i + 1] = (__bf16)f.y;
                As[arow][ahalf + 4*i + 2] = (__bf16)f.z;
                As[arow][ahalf + 4*i + 3] = (__bf16)f.w;
            }
        }
        // ---- stage B tile (32x128) transposed to Bs[n][k] ----
        {
            const F4* src = (const F4*)(B + (size_t)(kk + bk) * N + n0 + bn);
#pragma unroll
            for (int i = 0; i < 4; ++i) {
                F4 f = src[i];
                Bs[bn + 4*i + 0][bk] = (__bf16)f.x;
                Bs[bn + 4*i + 1][bk] = (__bf16)f.y;
                Bs[bn + 4*i + 2][bk] = (__bf16)f.z;
                Bs[bn + 4*i + 3][bk] = (__bf16)f.w;
            }
        }
        // ---- prefetch next K-step tiles (global_prefetch_b8) ----
        if (kk + 32 < K) {
            const size_t esz = A_BF16 ? 2 : 4;
            __builtin_prefetch((const char*)Ap +
                esz * ((size_t)(m0 + arow) * K + kk + 32 + ahalf), 0, 0);
            __builtin_prefetch(B + (size_t)(kk + 32 + bk) * N + n0 + bn, 0, 0);
        }
        __syncthreads();

        // ---- fragments + WMMA ----
        Frag a[2], b[4];
#pragma unroll
        for (int tm = 0; tm < 2; ++tm) {
            const int r = wm * 32 + tm * 16 + ln;
            a[tm].q[0] = *(const U4*)&As[r][klo];
            a[tm].q[1] = *(const U4*)&As[r][klo + 16];
        }
#pragma unroll
        for (int tn = 0; tn < 4; ++tn) {
            const int c = wn * 64 + tn * 16 + ln;
            b[tn].q[0] = *(const U4*)&Bs[c][klo];
            b[tn].q[1] = *(const U4*)&Bs[c][klo + 16];
        }
#pragma unroll
        for (int tm = 0; tm < 2; ++tm)
#pragma unroll
            for (int tn = 0; tn < 4; ++tn)
                acc[tm][tn] = __builtin_amdgcn_wmma_f32_16x16x32_bf16(
                    false, a[tm].v, false, b[tn].v, (short)0, acc[tm][tn],
                    false, false);
        __syncthreads();
    }

    // ---- epilogue: bias add, fp32 store ----
#pragma unroll
    for (int tm = 0; tm < 2; ++tm)
#pragma unroll
        for (int tn = 0; tn < 4; ++tn) {
            const int col = n0 + wn * 64 + tn * 16 + ln;
            const float bv = bias[col];
#pragma unroll
            for (int v = 0; v < 8; ++v) {
                const int row = m0 + wm * 32 + tm * 16 + v + rlo;
                C[(size_t)row * N + col] = acc[tm][tn][v] + bv;
            }
        }
}

// ---------------------------------------------------------------------------
// RoPE + head split.  qkv fp32 [b,t,3*1024] ->
//   Qh,Kh bf16 [b,h,t,64] (rope applied),  Vt bf16 [b,h,64,t] (transposed).
// ---------------------------------------------------------------------------
__global__ __launch_bounds__(256)
void rope_split_kernel(const float* __restrict__ qkv,
                       __bf16* __restrict__ Qh, __bf16* __restrict__ Kh,
                       __bf16* __restrict__ Vt, int T)
{
    const int idx = blockIdx.x * blockDim.x + threadIdx.x;
    const int i = idx & 31;
    const int h = (idx >> 5) & 15;
    const int t = (idx >> 9) % T;
    const int b = idx / (T << 9);

    const float* base = qkv + (size_t)(b * T + t) * 3072;
    const int d = h * 64 + 2 * i;
    const float qe = base[d],        qo = base[d + 1];
    const float ke = base[1024 + d], ko = base[1024 + d + 1];
    const float ve = base[2048 + d], vo = base[2048 + d + 1];

    const float inv = __powf(10000.f, -(2.f * (float)i) / 64.f);
    const float ang = (float)t * inv;
    const float c = __cosf(ang), s = __sinf(ang);

    const size_t bh = (size_t)(b * 16 + h);
    const size_t qi = (bh * T + t) * 64 + 2 * i;
    Qh[qi]     = (__bf16)(qe * c - qo * s);
    Qh[qi + 1] = (__bf16)(qo * c + qe * s);
    Kh[qi]     = (__bf16)(ke * c - ko * s);
    Kh[qi + 1] = (__bf16)(ko * c + ke * s);

    const size_t vi = (bh * 64 + 2 * i) * T + t;
    Vt[vi]     = (__bf16)ve;
    Vt[vi + T] = (__bf16)vo;
}

// ---------------------------------------------------------------------------
// Flash-style attention. Block = 4 waves; wave owns 16 query rows, hd=64.
// K/V tiles (32 keys) cooperatively staged into LDS with async-to-LDS copies,
// double buffered so the next tile's DMA overlaps this tile's WMMA/softmax.
// Row max via 16-lane shfl butterflies; row sum via P x ones WMMA (free in
// the C layout: every lane of the accumulator holds its row's sum).
// ---------------------------------------------------------------------------
__global__ __launch_bounds__(128)
void attn_kernel(const __bf16* __restrict__ Qh, const __bf16* __restrict__ Kh,
                 const __bf16* __restrict__ Vt, __bf16* __restrict__ AO, int T)
{
    __shared__ __bf16 Ks[2][32][72];    // [buf][key][hd]   pitch 144B
    __shared__ __bf16 Vs[2][64][40];    // [buf][d][key]    pitch 80B
    __shared__ __bf16 Pst[4][16][40];   // per-wave P tile (C-layout -> A-frag)

    const int tid  = threadIdx.x;
    const int lane = tid & 31;
    const int wave = tid >> 5;
    const int ln   = lane & 15;
    const int klo  = (lane < 16) ? 0 : 8;
    const int rlo  = (lane < 16) ? 0 : 8;
    const int h = blockIdx.y, b = blockIdx.z;
    const int q0 = blockIdx.x * 64 + wave * 16;
    const size_t bh = (size_t)(b * 16 + h);

    const __bf16* Qb = Qh + (bh * T + q0) * 64;
    const __bf16* Kb = Kh + bh * T * 64;
    const __bf16* Vb = Vt + bh * 64 * T;

    // staging work split: 128 threads x (2 K chunks + 2 V chunks) of 16B
    const int kRow = tid >> 2;            // 0..31 (key)
    const int kSeg = (tid & 3) * 16;      // hd element offset 0/16/32/48
    const int vRow = tid >> 1;            // 0..63 (d)
    const int vSeg = (tid & 1) * 16;      // key element offset 0/16

    // Q A-fragments (16x64 as 2 x 16x32), loaded once.
    Frag aq[2];
#pragma unroll
    for (int kc = 0; kc < 2; ++kc) {
        const __bf16* p = Qb + (size_t)ln * 64 + kc * 32 + klo;
        aq[kc].q[0] = *(const U4*)p;
        aq[kc].q[1] = *(const U4*)(p + 16);
    }

    // ones B-fragment for rowsum WMMA
    Frag ones;
#pragma unroll
    for (int i = 0; i < 16; ++i) ones.v[i] = (__bf16)1.0f;

    const v8f vz = {0.f,0.f,0.f,0.f,0.f,0.f,0.f,0.f};
    v8f o[4];
#pragma unroll
    for (int d = 0; d < 4; ++d) o[d] = vz;
    v8f rs = vz;                    // running row-sum l, same C layout as o
    float rmax[8];
#pragma unroll
    for (int v = 0; v < 8; ++v) rmax[v] = -3.0e38f;

    const float scale = 0.125f;     // 1/sqrt(64)
    const int nsteps = T / 32;

    // prologue: stage tile 0 into buffer 0
    {
        const __bf16* kg = Kb + (size_t)kRow * 64 + kSeg;
        copy16_g2l(kg,     &Ks[0][kRow][kSeg]);
        copy16_g2l(kg + 8, &Ks[0][kRow][kSeg + 8]);
        const __bf16* vg = Vb + (size_t)vRow * T + vSeg;
        copy16_g2l(vg,     &Vs[0][vRow][vSeg]);
        copy16_g2l(vg + 8, &Vs[0][vRow][vSeg + 8]);
    }

    for (int j = 0; j < nsteps; ++j) {
        const int buf = j & 1;
        if (j + 1 < nsteps) {
            const int j0n = (j + 1) * 32;
            const __bf16* kgn = Kb + (size_t)(j0n + kRow) * 64 + kSeg;
            copy16_g2l(kgn,     &Ks[buf ^ 1][kRow][kSeg]);
            copy16_g2l(kgn + 8, &Ks[buf ^ 1][kRow][kSeg + 8]);
            const __bf16* vgn = Vb + (size_t)vRow * T + j0n + vSeg;
            copy16_g2l(vgn,     &Vs[buf ^ 1][vRow][vSeg]);
            copy16_g2l(vgn + 8, &Vs[buf ^ 1][vRow][vSeg + 8]);
            async_wait_le4();           // previous tile complete
        } else {
            async_wait_0();
        }
        __syncthreads();                // staged data visible to all waves

        // ---- S = Q K^T for 2 key tiles (4 WMMA) ----
        Frag bk[2][2];
#pragma unroll
        for (int nt = 0; nt < 2; ++nt)
#pragma unroll
            for (int kc = 0; kc < 2; ++kc) {
                bk[nt][kc].q[0] = *(const U4*)&Ks[buf][nt * 16 + ln][kc * 32 + klo];
                bk[nt][kc].q[1] = *(const U4*)&Ks[buf][nt * 16 + ln][kc * 32 + klo + 16];
            }
        v8f s[2];
#pragma unroll
        for (int nt = 0; nt < 2; ++nt) {
            v8f t0 = __builtin_amdgcn_wmma_f32_16x16x32_bf16(
                false, aq[0].v, false, bk[nt][0].v, (short)0, vz, false, false);
            s[nt]  = __builtin_amdgcn_wmma_f32_16x16x32_bf16(
                false, aq[1].v, false, bk[nt][1].v, (short)0, t0, false, false);
        }

        // ---- online softmax: row max via 16-lane butterflies ----
#pragma unroll
        for (int v = 0; v < 8; ++v) {
            float a0 = s[0][v] * scale, a1 = s[1][v] * scale;
            float tm = fmaxf(a0, a1);
#pragma unroll
            for (int off = 1; off < 16; off <<= 1)
                tm = fmaxf(tm, __shfl_xor(tm, off, 32));
            const float mnew  = fmaxf(rmax[v], tm);
            const float alpha = __expf(rmax[v] - mnew);
            rmax[v] = mnew;
            const float p0 = __expf(a0 - mnew);
            const float p1 = __expf(a1 - mnew);
            rs[v] *= alpha;
#pragma unroll
            for (int d = 0; d < 4; ++d) o[d][v] *= alpha;
            Pst[wave][v + rlo][ln]      = (__bf16)p0;
            Pst[wave][v + rlo][16 + ln] = (__bf16)p1;
        }
        asm volatile("s_wait_dscnt 0" ::: "memory");   // P stores -> loads

        // P as A-fragment (16x32 over the 32 keys)
        Frag ap;
        ap.q[0] = *(const U4*)&Pst[wave][ln][klo];
        ap.q[1] = *(const U4*)&Pst[wave][ln][klo + 16];

        // rowsum: rs = P x ones + alpha*rs  (1 WMMA replaces 32 shuffles)
        rs = __builtin_amdgcn_wmma_f32_16x16x32_bf16(
            false, ap.v, false, ones.v, (short)0, rs, false, false);

        // ---- O += P V (4 WMMA) ----
        Frag bv[4];
#pragma unroll
        for (int dt = 0; dt < 4; ++dt) {
            bv[dt].q[0] = *(const U4*)&Vs[buf][dt * 16 + ln][klo];
            bv[dt].q[1] = *(const U4*)&Vs[buf][dt * 16 + ln][klo + 16];
        }
#pragma unroll
        for (int dt = 0; dt < 4; ++dt)
            o[dt] = __builtin_amdgcn_wmma_f32_16x16x32_bf16(
                false, ap.v, false, bv[dt].v, (short)0, o[dt], false, false);

        __syncthreads();                // all waves done with buf before reuse
    }

    // ---- epilogue: normalize, write bf16 AO [b,t,1024] ----
#pragma unroll
    for (int dt = 0; dt < 4; ++dt)
#pragma unroll
        for (int v = 0; v < 8; ++v) {
            const int tq = q0 + v + rlo;
            const int d  = h * 64 + dt * 16 + ln;
            AO[(size_t)(b * T + tq) * 1024 + d] = (__bf16)(o[dt][v] / rs[v]);
        }
}

// ---------------------------------------------------------------------------
// Launch
// ---------------------------------------------------------------------------
extern "C" void kernel_launch(void* const* d_in, const int* in_sizes, int n_in,
                              void* d_out, int out_size, void* d_ws, size_t ws_size,
                              hipStream_t stream)
{
    const float* x     = (const float*)d_in[0];
    const float* w_qkv = (const float*)d_in[1];
    const float* b_qkv = (const float*)d_in[2];
    const float* w_out = (const float*)d_in[3];
    const float* b_out = (const float*)d_in[4];
    float* out = (float*)d_out;

    const int B = 2, T = 2048, D = 1024, H = 16;
    const int M = B * T;                       // 4096

    char* ws = (char*)d_ws;
    float*  qkv = (float*)ws;                                  // 48 MB fp32
    __bf16* Qh  = (__bf16*)(ws + (size_t)M * 3 * D * sizeof(float));
    __bf16* Kh  = Qh + (size_t)M * D;                          // 8 MB each
    __bf16* Vt  = Kh + (size_t)M * D;
    __bf16* AO  = Vt + (size_t)M * D;

    // 1) QKV projection: [4096,1024] x [1024,3072] + bias
    gemm_bf16_wmma<false><<<dim3(3 * D / 128, M / 128), 256, 0, stream>>>(
        x, w_qkv, b_qkv, qkv, M, 3 * D, D);

    // 2) RoPE + head split (+ V transpose)
    rope_split_kernel<<<(M * D / 2) / 256, 256, 0, stream>>>(qkv, Qh, Kh, Vt, T);

    // 3) flash attention per (b, h, 64-query tile)
    attn_kernel<<<dim3(T / 64, H, B), 128, 0, stream>>>(Qh, Kh, Vt, AO, T);

    // 4) output projection: [4096,1024] x [1024,1024] + bias -> fp32 out
    gemm_bf16_wmma<true><<<dim3(D / 128, M / 128), 256, 0, stream>>>(
        AO, w_out, b_out, out, M, D, D);
}
